// my_gat_conv_71614284694253
// MI455X (gfx1250) — matrix operand order repaired and checked by reading, not compiled, not run on hardware
//
#include <hip/hip_runtime.h>

typedef __attribute__((ext_vector_type(16))) _Float16 v16h;
typedef __attribute__((ext_vector_type(2)))  _Float16 h2;
typedef __attribute__((ext_vector_type(8)))  float    v8f;
typedef int v4i __attribute__((vector_size(16)));
typedef __attribute__((address_space(1))) v4i* gv4i_p;   // global 16B-vector ptr
typedef __attribute__((address_space(3))) v4i* lv4i_p;   // LDS    16B-vector ptr

#define NNODES 50000
#define CH 96          // hidden/out channels
#define KMAX 256       // max K (layer 1)

#if defined(__HIP_DEVICE_COMPILE__) && __has_builtin(__builtin_amdgcn_global_load_async_to_lds_b128)
#define USE_ASYNC_LDS 1
#else
#define USE_ASYNC_LDS 0
#endif

// ---------------------------------------------------------------- utilities
__global__ void fill_kernel(float* p, long long n, float v) {
    long long i = (long long)blockIdx.x * blockDim.x + threadIdx.x;
    if (i < n) p[i] = v;
}

// out[i] = b[i % CH]  (pre-load aggregation target with bias)
__global__ void fill_bias_kernel(float* __restrict__ p, const float* __restrict__ b,
                                 long long n) {
    long long i = (long long)blockIdx.x * blockDim.x + threadIdx.x;
    if (i < n) p[i] = b[i % CH];
}

// W [K,N] fp32 row-major -> Wt [N,K] f16
__global__ void convert_wT_kernel(const float* __restrict__ W, _Float16* __restrict__ Wt,
                                  int K, int Nn) {
    int idx = blockIdx.x * blockDim.x + threadIdx.x;
    if (idx >= K * Nn) return;
    int k = idx / Nn, n = idx % Nn;
    Wt[(size_t)n * K + k] = (_Float16)W[idx];
}

// int64 edge_index [2,E] -> int32 src/dst arrays with self-loops appended
__global__ void prep_edges_kernel(const long long* __restrict__ ei, int nE, int nN,
                                  int* __restrict__ src32, int* __restrict__ dst32) {
    int e = blockIdx.x * blockDim.x + threadIdx.x;
    if (e >= nE + nN) return;
    if (e < nE) { src32[e] = (int)ei[e]; dst32[e] = (int)ei[(size_t)nE + e]; }
    else        { src32[e] = dst32[e] = e - nE; }
}

__device__ __forceinline__ float4 addrelu4(float4 a, float4 b) {
    float4 r;
    r.x = fmaxf(a.x + b.x, 0.f); r.y = fmaxf(a.y + b.y, 0.f);
    r.z = fmaxf(a.z + b.z, 0.f); r.w = fmaxf(a.w + b.w, 0.f);
    return r;
}

// ---------------------------------------------------------------- WMMA GEMM
// C[M,96] = act(A[M,K] + abias) * B[K,96]  (B as BT[96,K] f16), fused epilogue:
// s[m] = C[m,:]·avs, d[m] = C[m,:]·avd.  One wave computes 16 rows.
__global__ __launch_bounds__(128)
void gemm_wmma_kernel(const float* __restrict__ A, const _Float16* __restrict__ BT,
                      float* __restrict__ C, int M, int K,
                      const float* __restrict__ abias,        // null: identity input
                      const float* __restrict__ avs, const float* __restrict__ avd,
                      float* __restrict__ s, float* __restrict__ d) {
    __shared__ _Float16 BsT[CH * KMAX];   // 48 KB, well under 320 KB/WGP

    // ---- stage BT (contiguous 96*K halfs) into LDS
#if USE_ASYNC_LDS
    {
        gv4i_p g = (gv4i_p)BT;                    // AS(1) 16B-vector ptr
        lv4i_p l = (lv4i_p)BsT;                   // AS(3) 16B-vector ptr
        int nVec = (CH * K) >> 3;                 // 16-byte chunks
        for (int i = threadIdx.x; i < nVec; i += blockDim.x)
            __builtin_amdgcn_global_load_async_to_lds_b128(g + i, l + i, 0, 0);
#if __has_builtin(__builtin_amdgcn_s_wait_asynccnt)
        __builtin_amdgcn_s_wait_asynccnt(0);
#else
        asm volatile("s_wait_asynccnt 0" ::: "memory");
#endif
    }
#else
    {
        const unsigned* src = (const unsigned*)BT;
        unsigned* dst = (unsigned*)BsT;
        int nWords = (CH * K) >> 1;
        for (int i = threadIdx.x; i < nWords; i += blockDim.x) dst[i] = src[i];
    }
#endif
    __syncthreads();

    const int lane = threadIdx.x & 31;
    const int wave = threadIdx.x >> 5;
    const int m_base = (blockIdx.x * (blockDim.x >> 5) + wave) * 16;
    if (m_base >= M) return;                 // wave-uniform exit (after barrier)

    const int l16  = lane & 15;
    const int half = lane >> 4;              // 0: low half-wave, 1: high
    const int row  = m_base + l16;           // A-fragment: M = lane&15

    v8f acc[6];
#pragma unroll
    for (int t = 0; t < 6; ++t) acc[t] = (v8f){0.f,0.f,0.f,0.f,0.f,0.f,0.f,0.f};

    const int akoff = half * 8;              // A: half-wave K group offset
    for (int k0 = 0; k0 < K; k0 += 32) {
        // ---- A fragment (16-bit A 16x32 layout): VGPR j<4 -> K=akoff+2j..,
        //      VGPR j>=4 -> K=akoff+16+2(j-4)..
        const float* arow = A + (size_t)row * K + k0 + akoff;
        float4 a0 = *(const float4*)(arow + 0);
        float4 a1 = *(const float4*)(arow + 4);
        float4 a2 = *(const float4*)(arow + 16);
        float4 a3 = *(const float4*)(arow + 20);
        if (abias) {                          // fused relu(a + b) input transform
            const float* brow = abias + k0 + akoff;
            a0 = addrelu4(a0, *(const float4*)(brow + 0));
            a1 = addrelu4(a1, *(const float4*)(brow + 4));
            a2 = addrelu4(a2, *(const float4*)(brow + 16));
            a3 = addrelu4(a3, *(const float4*)(brow + 20));
        }
        v16h a;
        a[0]=(_Float16)a0.x; a[1]=(_Float16)a0.y; a[2]=(_Float16)a0.z; a[3]=(_Float16)a0.w;
        a[4]=(_Float16)a1.x; a[5]=(_Float16)a1.y; a[6]=(_Float16)a1.z; a[7]=(_Float16)a1.w;
        a[8]=(_Float16)a2.x; a[9]=(_Float16)a2.y; a[10]=(_Float16)a2.z; a[11]=(_Float16)a2.w;
        a[12]=(_Float16)a3.x; a[13]=(_Float16)a3.y; a[14]=(_Float16)a3.z; a[15]=(_Float16)a3.w;

        const int kb = k0 + half * 8;        // B: half-wave K group offset (matches A)
#pragma unroll
        for (int t = 0; t < 6; ++t) {
            const int n = t * 16 + l16;      // B-fragment: N = lane&15
            const _Float16* bp = BsT + (size_t)n * K + kb;
            v16h b;
#pragma unroll
            for (int j = 0; j < 8; ++j) {    // VGPR j<4 -> K=kb+2j.., j>=4 -> K=kb+16+2(j-4)..
                h2 p = __builtin_bit_cast(h2, *(const unsigned*)(bp + 2 * j + (j >= 4 ? 8 : 0)));
                b[2*j]   = p[0];
                b[2*j+1] = p[1];
            }
            acc[t] = __builtin_amdgcn_wmma_f32_16x16x32_f16(
                false, a, false, b, (short)0, acc[t], false, false);
        }
    }

    // ---- store h: C/D layout: lane -> N=l16, VGPR r -> M = half*8 + r
#pragma unroll
    for (int t = 0; t < 6; ++t) {
        float* cp = C + (size_t)(m_base + half * 8) * CH + t * 16 + l16;
#pragma unroll
        for (int r = 0; r < 8; ++r) cp[(size_t)r * CH] = acc[t][r];
    }

    // ---- fused attention projections: s/d[m] = sum_n C[m,n]*av[n]
    float av[6], dv[6];
#pragma unroll
    for (int t = 0; t < 6; ++t) { av[t] = avs[t*16 + l16]; dv[t] = avd[t*16 + l16]; }
#pragma unroll
    for (int r = 0; r < 8; ++r) {
        float sp = 0.f, dp = 0.f;
#pragma unroll
        for (int t = 0; t < 6; ++t) { sp += acc[t][r] * av[t]; dp += acc[t][r] * dv[t]; }
#pragma unroll
        for (int off = 1; off < 16; off <<= 1) {   // reduce within each 16-lane half
            sp += __shfl_xor(sp, off, 32);
            dp += __shfl_xor(dp, off, 32);
        }
        if (l16 == 0) {
            int m = m_base + half * 8 + r;
            s[m] = sp; d[m] = dp;
        }
    }
}

// ---------------------------------------------------------------- edge phase
__device__ __forceinline__ void atomicMaxFloat(float* addr, float v) {
    if (v >= 0.f) atomicMax((int*)addr, __float_as_int(v));
    else          atomicMin((unsigned int*)addr, __float_as_uint(v));
}

__global__ void edge_max_kernel(const int* __restrict__ src32, const int* __restrict__ dst32,
                                int Et, const float* __restrict__ s,
                                const float* __restrict__ d,
                                float* __restrict__ escore, float* __restrict__ emax) {
    int e = blockIdx.x * blockDim.x + threadIdx.x;
    if (e >= Et) return;
    int dst = dst32[e];
    float sc = s[src32[e]] + d[dst];
    sc = sc > 0.f ? sc : 0.2f * sc;            // leaky_relu(., 0.2)
    escore[e] = sc;
    atomicMaxFloat(emax + dst, sc);
}

__global__ void edge_exp_kernel(const int* __restrict__ dst32, int Et,
                                const float* __restrict__ emax,
                                float* __restrict__ escore, float* __restrict__ denom) {
    int e = blockIdx.x * blockDim.x + threadIdx.x;
    if (e >= Et) return;
    int dst = dst32[e];
    float ee = __expf(escore[e] - emax[dst]);
    escore[e] = ee;
    atomicAdd(denom + dst, ee);
}

// one wave per edge: 96 channels in 3 strides of 32
__global__ void aggregate_kernel(const int* __restrict__ src32, const int* __restrict__ dst32,
                                 int Et, const float* __restrict__ h,
                                 const float* __restrict__ escore,
                                 const float* __restrict__ denom, float* __restrict__ out) {
    int e = blockIdx.x * (blockDim.x >> 5) + (threadIdx.x >> 5);
    if (e >= Et) return;
    int lane = threadIdx.x & 31;
    int src = src32[e], dst = dst32[e];
    float w = escore[e] / (denom[dst] + 1e-16f);
    const float* hp = h + (size_t)src * CH;
    float* op = out + (size_t)dst * CH;
    atomicAdd(op + lane,      hp[lane]      * w);
    atomicAdd(op + lane + 32, hp[lane + 32] * w);
    atomicAdd(op + lane + 64, hp[lane + 64] * w);
}

// ------------------------------------------------------------------ launcher
extern "C" void kernel_launch(void* const* d_in, const int* in_sizes, int n_in,
                              void* d_out, int out_size, void* d_ws, size_t ws_size,
                              hipStream_t stream) {
    const float*     x   = (const float*)d_in[0];
    const long long* ei  = (const long long*)d_in[1];   // int64 edge_index [2,E]
    const float*     W1  = (const float*)d_in[3];
    const float*     as1 = (const float*)d_in[4];
    const float*     ad1 = (const float*)d_in[5];
    const float*     b1  = (const float*)d_in[6];
    const float*     W2  = (const float*)d_in[7];
    const float*     as2 = (const float*)d_in[8];
    const float*     ad2 = (const float*)d_in[9];
    const float*     b2  = (const float*)d_in[10];
    float* out = (float*)d_out;

    const int N  = NNODES;
    const int E  = in_sizes[1] / 2;
    const int Et = E + N;
    const long long HN = (long long)N * CH;

    // workspace layout
    char* ws = (char*)d_ws;
    float* hbuf   = (float*)ws; ws += HN * 4;
    float* aggbuf = (float*)ws; ws += HN * 4;
    float* sbuf   = (float*)ws; ws += (size_t)N * 4;
    float* dbuf   = (float*)ws; ws += (size_t)N * 4;
    float* emax   = (float*)ws; ws += (size_t)N * 4;
    float* denom  = (float*)ws; ws += (size_t)N * 4;
    float* escore = (float*)ws; ws += (size_t)Et * 4;
    int*   src32  = (int*)ws;   ws += (size_t)Et * 4;
    int*   dst32  = (int*)ws;   ws += (size_t)Et * 4;
    _Float16* w1t = (_Float16*)ws; ws += (size_t)256 * CH * 2;
    _Float16* w2t = (_Float16*)ws;

    const int TB = 256;
    auto cdiv = [](long long a, long long b) { return (int)((a + b - 1) / b); };

    // one-time prep
    convert_wT_kernel<<<cdiv(256 * CH, TB), TB, 0, stream>>>(W1, w1t, 256, CH);
    convert_wT_kernel<<<cdiv(CH * CH, TB), TB, 0, stream>>>(W2, w2t, CH, CH);
    prep_edges_kernel<<<cdiv(Et, TB), TB, 0, stream>>>(ei, E, N, src32, dst32);

    const int gemmBlocks = cdiv(N, 64);      // 128 threads = 4 waves = 64 rows
    const int edgeBlocks = cdiv(Et, TB);
    const int aggBlocks  = cdiv(Et, TB / 32);

    // -------- layer 1: h1 = x@W1 (+ fused s1,d1)
    gemm_wmma_kernel<<<gemmBlocks, 128, 0, stream>>>(x, w1t, hbuf, N, 256,
                                                     nullptr, as1, ad1, sbuf, dbuf);
    fill_kernel<<<cdiv(N, TB), TB, 0, stream>>>(emax, N, -3.0e38f);
    fill_kernel<<<cdiv(N, TB), TB, 0, stream>>>(denom, N, 0.f);
    edge_max_kernel<<<edgeBlocks, TB, 0, stream>>>(src32, dst32, Et, sbuf, dbuf, escore, emax);
    edge_exp_kernel<<<edgeBlocks, TB, 0, stream>>>(dst32, Et, emax, escore, denom);
    fill_kernel<<<cdiv(HN, TB), TB, 0, stream>>>(aggbuf, HN, 0.f);
    aggregate_kernel<<<aggBlocks, TB, 0, stream>>>(src32, dst32, Et, hbuf, escore, denom, aggbuf);

    // -------- layer 2: h2 = relu(agg1 + b1)@W2 (input fused) (+ s2,d2)
    gemm_wmma_kernel<<<gemmBlocks, 128, 0, stream>>>(aggbuf, w2t, hbuf, N, CH,
                                                     b1, as2, ad2, sbuf, dbuf);
    fill_kernel<<<cdiv(N, TB), TB, 0, stream>>>(emax, N, -3.0e38f);
    fill_kernel<<<cdiv(N, TB), TB, 0, stream>>>(denom, N, 0.f);
    edge_max_kernel<<<edgeBlocks, TB, 0, stream>>>(src32, dst32, Et, sbuf, dbuf, escore, emax);
    edge_exp_kernel<<<edgeBlocks, TB, 0, stream>>>(dst32, Et, emax, escore, denom);
    fill_bias_kernel<<<cdiv(HN, TB), TB, 0, stream>>>(out, b2, HN);   // init with b2
    aggregate_kernel<<<aggBlocks, TB, 0, stream>>>(src32, dst32, Et, hbuf, escore, denom, out);
}
